// GCN_74663711473765
// MI455X (gfx1250) — matrix operand order repaired and checked by reading, not compiled, not run on hardware
//
#include <hip/hip_runtime.h>
#include <hip/hip_bf16.h>

typedef __attribute__((ext_vector_type(16))) __bf16 v16bf;
typedef __attribute__((ext_vector_type(8)))  __bf16 v8bf;
typedef __attribute__((ext_vector_type(4)))  __bf16 v4bf;
typedef __attribute__((ext_vector_type(8)))  float  v8f;

#define NN     98304      // nodes
#define EE     196608     // edges (without self loops)
#define BB     4096       // graphs
#define MMQ    32768      // B*K cliques
#define EMB_D  300
#define FEAT_D 512
#define EPSF   1e-5f

static __device__ __forceinline__ float softplusf(float x) {
  return fmaxf(x, 0.f) + log1pf(expf(-fabsf(x)));  // stable softplus
}

// ---------------------------------------------------------------------------
// Generic f32->bf16 WMMA GEMM: C[MxN] = act(A[MxK] @ B[KxN] + bias + res)
// 256 threads (8 waves), 128x64 block tile, BK=64 (2 WMMA k-steps per stage),
// wave tile 32x32 -> 8 static v_wmma_f32_16x16x32_bf16 per stage.
// Fast vectorized staging (b128 loads, b64/b128 LDS stores) for interior
// tiles; guarded scalar path only on edges. act: 0=none, 1=relu, 2=softplus
// ---------------------------------------------------------------------------
__global__ __launch_bounds__(256)
void gemm_bf16_wmma(const float* __restrict__ A, const float* __restrict__ B,
                    const float* __restrict__ bias, const float* __restrict__ res,
                    float* __restrict__ C, int M, int N, int K, int act)
{
  __shared__ alignas(64) __bf16 sA[128][64];   // [row][k]        16 KB
  __shared__ alignas(64) __bf16 sB[64][64];    // transposed [col][k]  8 KB

  const int tid  = threadIdx.x;
  const int lane = tid & 31;
  const int wave = tid >> 5;
  const int wm   = (wave >> 1) * 32;          // wave sub-tile in M (0..96)
  const int wn   = (wave & 1) * 32;           // wave sub-tile in N (0/32)
  const int m0   = blockIdx.y * 128;
  const int n0   = blockIdx.x * 64;
  const int g    = lane >> 4;                 // lane half (0/1)
  const int m16  = lane & 15;

  v8f acc[2][2] = {};

  for (int k0 = 0; k0 < K; k0 += 64) {
    // speculative prefetch of next A tile -> global_prefetch_b8
    if (k0 + 64 < K) {
      int pr = m0 + (tid >> 1);
      __builtin_prefetch(&A[(long)pr * K + k0 + 64], 0, 0);
    }

    // ---- stage A tile (128 x 64) ----
    if (k0 + 64 <= K) {                        // M is always a mult of 128 here
      int r     = tid >> 1;                    // 0..127
      int halfc = (tid & 1) * 32;              // 0 / 32
      const float4* ap = (const float4*)&A[(long)(m0 + r) * K + k0 + halfc];
#pragma unroll
      for (int q = 0; q < 4; ++q) {
        float4 fa = ap[2 * q], fb = ap[2 * q + 1];
        v8bf o;
        o[0] = (__bf16)fa.x; o[1] = (__bf16)fa.y; o[2] = (__bf16)fa.z; o[3] = (__bf16)fa.w;
        o[4] = (__bf16)fb.x; o[5] = (__bf16)fb.y; o[6] = (__bf16)fb.z; o[7] = (__bf16)fb.w;
        *(v8bf*)&sA[r][halfc + q * 8] = o;
      }
    } else {                                   // K tail (K=300 GEMMs)
#pragma unroll
      for (int i = 0; i < 32; ++i) {
        int idx = i * 256 + tid;
        int r   = idx >> 6;
        int kk  = idx & 63;
        float v = 0.f;
        if ((m0 + r) < M && (k0 + kk) < K) v = A[(long)(m0 + r) * K + (k0 + kk)];
        sA[r][kk] = (__bf16)v;
      }
    }

    // ---- stage B tile (64 x 64), transposed into [col][k] ----
    if (k0 + 64 <= K && n0 + 64 <= N) {
      int kk4 = (tid & 15) * 4;                // 0..60
      int c4  = (tid >> 4) * 4;                // 0..60
      const float* bp = &B[(long)(k0 + kk4) * N + n0 + c4];
      float4 r0 = *(const float4*)(bp);
      float4 r1 = *(const float4*)(bp + N);
      float4 r2 = *(const float4*)(bp + 2 * (long)N);
      float4 r3 = *(const float4*)(bp + 3 * (long)N);
      v4bf t;
      t[0]=(__bf16)r0.x; t[1]=(__bf16)r1.x; t[2]=(__bf16)r2.x; t[3]=(__bf16)r3.x;
      *(v4bf*)&sB[c4 + 0][kk4] = t;
      t[0]=(__bf16)r0.y; t[1]=(__bf16)r1.y; t[2]=(__bf16)r2.y; t[3]=(__bf16)r3.y;
      *(v4bf*)&sB[c4 + 1][kk4] = t;
      t[0]=(__bf16)r0.z; t[1]=(__bf16)r1.z; t[2]=(__bf16)r2.z; t[3]=(__bf16)r3.z;
      *(v4bf*)&sB[c4 + 2][kk4] = t;
      t[0]=(__bf16)r0.w; t[1]=(__bf16)r1.w; t[2]=(__bf16)r2.w; t[3]=(__bf16)r3.w;
      *(v4bf*)&sB[c4 + 3][kk4] = t;
    } else {
#pragma unroll
      for (int i = 0; i < 16; ++i) {
        int idx = i * 256 + tid;
        int kk  = idx >> 6;
        int c   = idx & 63;
        float v = 0.f;
        if ((k0 + kk) < K && (n0 + c) < N) v = B[(long)(k0 + kk) * N + (n0 + c)];
        sB[c][kk] = (__bf16)v;
      }
    }
    __syncthreads();

    // ---- 2 WMMA k-steps over the staged 64-deep tile ----
#pragma unroll
    for (int ks = 0; ks < 2; ++ks) {
      const int koff = ks * 32;
      // A frag: lane<16 holds K 0..15 of row m16; lane>=16 holds K 16..31
      v16bf af[2], bfr[2];
#pragma unroll
      for (int t = 0; t < 2; ++t) {
        af[t] = *(const v16bf*)&sA[wm + t * 16 + m16][koff + g * 16];
      }
      // B frag: lane<16 holds K 0..15 of col m16; lane>=16 holds K 16..31
#pragma unroll
      for (int t = 0; t < 2; ++t)
        bfr[t] = *(const v16bf*)&sB[wn + t * 16 + m16][koff + g * 16];

#pragma unroll
      for (int i = 0; i < 2; ++i)
#pragma unroll
        for (int j = 0; j < 2; ++j)
          acc[i][j] = __builtin_amdgcn_wmma_f32_16x16x32_bf16(
              false, af[i], false, bfr[j], (short)0, acc[i][j], false, false);
    }
    __syncthreads();
  }

  // C/D layout: VGPR r -> M = g*8 + r (within 16), N = m16
#pragma unroll
  for (int i = 0; i < 2; ++i)
#pragma unroll
    for (int j = 0; j < 2; ++j) {
      int col = n0 + wn + j * 16 + m16;
      if (col >= N) continue;
#pragma unroll
      for (int r = 0; r < 8; ++r) {
        int row = m0 + wm + i * 16 + g * 8 + r;
        if (row >= M) continue;
        float v = acc[i][j][r];
        if (bias) v += bias[col];
        if (res)  v += res[(long)row * N + col];
        if (act == 1)      v = fmaxf(v, 0.f);
        else if (act == 2) v = softplusf(v);
        C[(long)row * N + col] = v;
      }
    }
}

// ---------------------------------------------------------------------------
// Elementwise / graph kernels
// ---------------------------------------------------------------------------
__global__ void embed_kernel(const int* __restrict__ x, const float* __restrict__ e1,
                             const float* __restrict__ e2, float* __restrict__ h)
{
  long i = (long)blockIdx.x * blockDim.x + threadIdx.x;
  if (i >= (long)NN * EMB_D) return;
  int n = (int)(i / EMB_D), c = (int)(i % EMB_D);
  h[i] = e1[x[2 * n] * EMB_D + c] + e2[x[2 * n + 1] * EMB_D + c];
}

__global__ void fill_bias_kernel(float* __restrict__ agg, const float* __restrict__ b)
{
  long i = (long)blockIdx.x * blockDim.x + threadIdx.x;
  if (i >= (long)NN * EMB_D) return;
  agg[i] = b[i % EMB_D];
}

// msg = hx[src] + (ee1[bond] + ee2[dir]);  agg[dst] += msg   (incl. self loops)
__global__ void scatter_kernel(const float* __restrict__ hx, const int* __restrict__ ei,
                               const int* __restrict__ ea, const float* __restrict__ ee1,
                               const float* __restrict__ ee2, float* __restrict__ agg)
{
  long i = (long)blockIdx.x * blockDim.x + threadIdx.x;
  if (i >= (long)(EE + NN) * EMB_D) return;
  int e = (int)(i / EMB_D), c = (int)(i % EMB_D);
  int s, d; float emb;
  if (e < EE) { s = ei[e]; d = ei[EE + e]; emb = ee1[ea[2 * e]] + ee2[ea[2 * e + 1]]; }
  else        { s = d = e - EE;            emb = ee1[4] + ee2[0]; }   // self-loop
  atomicAdd(&agg[(long)d * EMB_D + c], hx[(long)s * EMB_D + c] + emb);
}

__global__ __launch_bounds__(256)
void bn_stats_kernel(const float* __restrict__ agg, float* __restrict__ stats)
{
  __shared__ float r1[256], r2[256];
  int c = blockIdx.x, tid = threadIdx.x;
  float s = 0.f, s2 = 0.f;
  for (int n = tid; n < NN; n += 256) {
    float v = agg[(long)n * EMB_D + c];
    s += v; s2 += v * v;
  }
  r1[tid] = s; r2[tid] = s2; __syncthreads();
  for (int st = 128; st > 0; st >>= 1) {
    if (tid < st) { r1[tid] += r1[tid + st]; r2[tid] += r2[tid + st]; }
    __syncthreads();
  }
  if (tid == 0) {
    float mu  = r1[0] / (float)NN;
    float var = r2[0] / (float)NN - mu * mu;
    stats[c]         = mu;
    stats[EMB_D + c] = rsqrtf(var + EPSF);
  }
}

__global__ void bn_apply_kernel(const float* __restrict__ agg, const float* __restrict__ stats,
                                const float* __restrict__ g, const float* __restrict__ b,
                                float* __restrict__ h, int relu)
{
  long i = (long)blockIdx.x * blockDim.x + threadIdx.x;
  if (i >= (long)NN * EMB_D) return;
  int c = (int)(i % EMB_D);
  float v = (agg[i] - stats[c]) * stats[EMB_D + c] * g[c] + b[c];
  h[i] = relu ? fmaxf(v, 0.f) : v;
}

// batch = repeat(arange(B),24): contiguous 24-node segments -> mean pool
__global__ void pool_kernel(const float* __restrict__ h, float* __restrict__ hgp)
{
  long i = (long)blockIdx.x * blockDim.x + threadIdx.x;
  if (i >= (long)BB * EMB_D) return;
  int bq = (int)(i / EMB_D), c = (int)(i % EMB_D);
  float s = 0.f;
  for (int j = 0; j < 24; ++j) s += h[((long)bq * 24 + j) * EMB_D + c];
  hgp[i] = s * (1.f / 24.f);
}

__global__ void gather_kernel(const float* __restrict__ emb, const int* __restrict__ idx,
                              float* __restrict__ hp)
{
  long i = (long)blockIdx.x * blockDim.x + threadIdx.x;
  if (i >= (long)MMQ * FEAT_D) return;
  int m = (int)(i / FEAT_D), c = (int)(i % FEAT_D);
  hp[i] = emb[(long)idx[m] * FEAT_D + c];
}

__global__ __launch_bounds__(256)
void layernorm_kernel(float* __restrict__ x, const float* __restrict__ g,
                      const float* __restrict__ b)
{
  __shared__ float red[256];
  int row = blockIdx.x, tid = threadIdx.x;
  float* p = x + (long)row * FEAT_D;
  float v0 = p[tid], v1 = p[tid + 256];
  red[tid] = v0 + v1; __syncthreads();
  for (int st = 128; st > 0; st >>= 1) { if (tid < st) red[tid] += red[tid + st]; __syncthreads(); }
  float mu = red[0] * (1.f / FEAT_D);
  __syncthreads();
  float d0 = v0 - mu, d1 = v1 - mu;
  red[tid] = d0 * d0 + d1 * d1; __syncthreads();
  for (int st = 128; st > 0; st >>= 1) { if (tid < st) red[tid] += red[tid + st]; __syncthreads(); }
  float rstd = rsqrtf(red[0] * (1.f / FEAT_D) + EPSF);
  p[tid]       = d0 * rstd * g[tid]       + b[tid];
  p[tid + 256] = d1 * rstd * g[tid + 256] + b[tid + 256];
}

// softmax over singleton q-axis ==> A==1 ==> attention out = sum_k V
__global__ void vsum_kernel(const float* __restrict__ vn, float* __restrict__ attn)
{
  long i = (long)blockIdx.x * blockDim.x + threadIdx.x;
  if (i >= (long)BB * FEAT_D) return;
  int bq = (int)(i / FEAT_D), c = (int)(i % FEAT_D);
  float s = 0.f;
  for (int k = 0; k < 8; ++k) s += vn[((long)bq * 8 + k) * FEAT_D + c];
  attn[i] = s;
}

// ---------------------------------------------------------------------------
extern "C" void kernel_launch(void* const* d_in, const int* in_sizes, int n_in,
                              void* d_out, int out_size, void* d_ws, size_t ws_size,
                              hipStream_t stream)
{
  (void)in_sizes; (void)n_in; (void)out_size; (void)ws_size;

  const int*   x          = (const int*)  d_in[0];
  const int*   edge_index = (const int*)  d_in[1];
  const int*   edge_attr  = (const int*)  d_in[2];
  const int*   clique_idx = (const int*)  d_in[4];
  const float* x_emb1     = (const float*)d_in[6];
  const float* x_emb2     = (const float*)d_in[7];
  const float* gnn_W      = (const float*)d_in[8];
  const float* gnn_b      = (const float*)d_in[9];
  const float* ee1        = (const float*)d_in[10];
  const float* ee2        = (const float*)d_in[11];
  const float* bn_g       = (const float*)d_in[12];
  const float* bn_b       = (const float*)d_in[13];
  const float* feat_W     = (const float*)d_in[14];
  const float* feat_b     = (const float*)d_in[15];
  const float* clique_emb = (const float*)d_in[16];
  const float* enc_W      = (const float*)d_in[17];
  const float* enc_b      = (const float*)d_in[18];
  const float* ln_g       = (const float*)d_in[19];
  const float* ln_b       = (const float*)d_in[20];
  const float* v_W        = (const float*)d_in[25];
  const float* v_b        = (const float*)d_in[26];
  const float* o_W        = (const float*)d_in[27];
  const float* o_b        = (const float*)d_in[28];
  const float* f1_W       = (const float*)d_in[29];
  const float* f1_b       = (const float*)d_in[30];
  const float* f2_W       = (const float*)d_in[31];
  const float* f2_b       = (const float*)d_in[32];
  const float* dec_W      = (const float*)d_in[33];
  const float* dec_b      = (const float*)d_in[34];
  const float* p1_W       = (const float*)d_in[35];
  const float* p1_b       = (const float*)d_in[36];
  const float* p2_W       = (const float*)d_in[37];
  const float* p2_b       = (const float*)d_in[38];
  const float* p3_W       = (const float*)d_in[39];
  const float* p3_b       = (const float*)d_in[40];

  float* ws = (float*)d_ws;
  const long NB = (long)NN * EMB_D;
  float* h   = ws;                 // node features
  float* hx  = ws + NB;            // h @ W
  float* agg = ws + 2 * NB;        // scattered aggregate
  float* sm  = ws + 3 * NB;
  float* stats  = sm;  sm += 1024;
  float* hg_pre = sm;  sm += (long)BB * EMB_D;
  float* hg     = sm;  sm += (long)BB * FEAT_D;
  float* attn   = sm;  sm += (long)BB * FEAT_D;
  float* out1   = sm;  sm += (long)BB * FEAT_D;
  float* f1t    = sm;  sm += (long)BB * FEAT_D;
  float* out2   = sm;  sm += (long)BB * FEAT_D;
  float* hm     = sm;  sm += (long)BB * FEAT_D;
  float* z1     = sm;  sm += (long)BB * 256;
  float* zsb    = sm;  sm += (long)BB * 256;
  float* z2     = sm;  sm += (long)BB * 256;
  // after pooling the 3 big node buffers are dead -> reuse for motif stage
  float* hp    = h;    // M x 512
  float* dense = hx;   // M x 512
  float* vn    = agg;  // M x 512

  auto ew = [](long n) { return dim3((unsigned)((n + 255) / 256)); };
  auto gr = [](int M, int N) { return dim3((unsigned)((N + 63) / 64),
                                           (unsigned)((M + 127) / 128)); };

  embed_kernel<<<ew(NB), 256, 0, stream>>>(x, x_emb1, x_emb2, h);

  for (int l = 0; l < 5; ++l) {
    gemm_bf16_wmma<<<gr(NN, EMB_D), 256, 0, stream>>>(
        h, gnn_W + (long)l * EMB_D * EMB_D, nullptr, nullptr, hx, NN, EMB_D, EMB_D, 0);
    fill_bias_kernel<<<ew(NB), 256, 0, stream>>>(agg, gnn_b + l * EMB_D);
    scatter_kernel<<<ew((long)(EE + NN) * EMB_D), 256, 0, stream>>>(
        hx, edge_index, edge_attr, ee1 + l * 5, ee2 + l * 3, agg);
    bn_stats_kernel<<<EMB_D, 256, 0, stream>>>(agg, stats);
    bn_apply_kernel<<<ew(NB), 256, 0, stream>>>(agg, stats, bn_g + l * EMB_D,
                                                bn_b + l * EMB_D, h, (l < 4) ? 1 : 0);
  }

  pool_kernel<<<ew((long)BB * EMB_D), 256, 0, stream>>>(h, hg_pre);
  gemm_bf16_wmma<<<gr(BB, FEAT_D), 256, 0, stream>>>(hg_pre, feat_W, feat_b, nullptr,
                                                     hg, BB, FEAT_D, EMB_D, 0);

  gather_kernel<<<ew((long)MMQ * FEAT_D), 256, 0, stream>>>(clique_emb, clique_idx, hp);
  gemm_bf16_wmma<<<gr(MMQ, FEAT_D), 256, 0, stream>>>(hp, enc_W, enc_b, nullptr,
                                                      dense, MMQ, FEAT_D, FEAT_D, 0);
  layernorm_kernel<<<MMQ, 256, 0, stream>>>(dense, ln_g, ln_b);
  gemm_bf16_wmma<<<gr(MMQ, FEAT_D), 256, 0, stream>>>(dense, v_W, v_b, nullptr,
                                                      vn, MMQ, FEAT_D, FEAT_D, 0);
  vsum_kernel<<<ew((long)BB * FEAT_D), 256, 0, stream>>>(vn, attn);

  gemm_bf16_wmma<<<gr(BB, FEAT_D), 256, 0, stream>>>(attn, o_W, o_b, hg,    out1, BB, FEAT_D, FEAT_D, 0);
  gemm_bf16_wmma<<<gr(BB, FEAT_D), 256, 0, stream>>>(out1, f1_W, f1_b, nullptr, f1t, BB, FEAT_D, FEAT_D, 1);
  gemm_bf16_wmma<<<gr(BB, FEAT_D), 256, 0, stream>>>(f1t,  f2_W, f2_b, out1,  out2, BB, FEAT_D, FEAT_D, 0);
  gemm_bf16_wmma<<<gr(BB, FEAT_D), 256, 0, stream>>>(out2, dec_W, dec_b, nullptr, hm, BB, FEAT_D, FEAT_D, 0);

  // hc = [hg | hm]  ->  split p1 into two GEMMs (second accumulates + softplus)
  gemm_bf16_wmma<<<gr(BB, 256), 256, 0, stream>>>(hg, p1_W,             p1_b,    nullptr, z1,  BB, 256, FEAT_D, 0);
  gemm_bf16_wmma<<<gr(BB, 256), 256, 0, stream>>>(hm, p1_W + 512 * 256, nullptr, z1,      zsb, BB, 256, FEAT_D, 2);
  gemm_bf16_wmma<<<gr(BB, 256), 256, 0, stream>>>(zsb, p2_W, p2_b, nullptr, z2, BB, 256, 256, 2);

  gemm_bf16_wmma<<<gr(BB, 2), 256, 0, stream>>>(z2, p3_W, p3_b, nullptr,
                                                (float*)d_out, BB, 2, 256, 0);
}
